// Decoder_15676630630950
// MI455X (gfx1250) — matrix operand order repaired and checked by reading, not compiled, not run on hardware
//
#include <hip/hip_runtime.h>

typedef __bf16 bf16;
typedef __attribute__((ext_vector_type(16))) __bf16 v16bf;
typedef __attribute__((ext_vector_type(8)))  __bf16 v8bf;
typedef __attribute__((ext_vector_type(8)))  float  v8f;

#define HD   512
#define ENCD 1024
#define VSZ  10000
#define BB   32
#define TT   500
#define LL   100
#define H4   2048
#define KX   1536          // HD + ENCD (concatenated [s|g] activation)
#define NEGINF (-1e10f)

__device__ __forceinline__ float sigmoidf(float x) { return 1.f / (1.f + __expf(-x)); }

// ---------------------------------------------------------------------------
// Core: one 16x16 f32 tile = A[16xK] (row-major, bf16) x Wt[NxK] (W transposed,
// row-major bf16, i.e. B column-contiguous).  Follows CDNA5 wave32 VGPR layouts:
//   A 16x32 bf16 : lanes 0-15 -> K {k0..k0+7, k0+16..k0+23}; lanes 16-31 -> +8
//   B 32x16 bf16 : same K mapping as A (lanes 0-15 -> {k0..k0+7, k0+16..k0+23}; lanes 16-31 -> +8)
// All loads are contiguous 16B (global_load_b128).
// ---------------------------------------------------------------------------
__device__ __forceinline__ v8f wmma_tile_bf16(const bf16* __restrict__ A, int lda,
                                              const bf16* __restrict__ Wt, int ldw,
                                              int K, int mBase, int nBase) {
  const int lane = threadIdx.x & 31;
  const int hf   = lane >> 4;        // 0: lanes 0-15, 1: lanes 16-31
  const int l16  = lane & 15;
  const bf16* __restrict__ Arow = A  + (size_t)(mBase + l16) * lda;
  const bf16* __restrict__ Wcol = Wt + (size_t)(nBase + l16) * ldw;
  v8f acc = {};
  for (int k0 = 0; k0 < K; k0 += 32) {
    v8bf a0 = *(const v8bf*)(Arow + k0 + (hf ? 8 : 0));
    v8bf a1 = *(const v8bf*)(Arow + k0 + 16 + (hf ? 8 : 0));
    v8bf b0 = *(const v8bf*)(Wcol + k0 + (hf ? 8 : 0));
    v8bf b1 = *(const v8bf*)(Wcol + k0 + 16 + (hf ? 8 : 0));
    v16bf a, b;
#pragma unroll
    for (int e = 0; e < 8; ++e) {
      a[e] = a0[e]; a[8 + e] = a1[e];
      b[e] = b0[e]; b[8 + e] = b1[e];
    }
    acc = __builtin_amdgcn_wmma_f32_16x16x32_bf16(false, a, false, b,
                                                  (short)0, acc, false, false);
  }
  return acc;
}

// ------------------------------- setup kernels -----------------------------
__global__ void k_cvt(const float* __restrict__ s, bf16* __restrict__ d, int n) {
  int i = blockIdx.x * blockDim.x + threadIdx.x;
  if (i < n) d[i] = (bf16)s[i];
}

// dst[n*ldd + koff + k] = src[k*N + n]   (fp32 -> bf16, transpose)
__global__ void k_cvtT(const float* __restrict__ src, bf16* __restrict__ dst,
                       int K, int N, int ldd, int koff) {
  int i = blockIdx.x * blockDim.x + threadIdx.x;
  if (i >= K * N) return;
  int k = i / N, n = i - k * N;
  dst[(size_t)n * ldd + koff + k] = (bf16)src[i];
}

__global__ void k_init(float* __restrict__ c, bf16* __restrict__ xg) {
  int i = blockIdx.x * blockDim.x + threadIdx.x;
  if (i < BB * HD) c[i] = 0.f;
  if (i < BB * KX) xg[i] = (bf16)0.f;
}

// h_proj[b*T+t, :HD] = bf16(h_batch @ attn_Wh + attn_b)
__global__ void k_hproj(const bf16* __restrict__ h_bf, const bf16* __restrict__ WhT,
                        const float* __restrict__ attn_b, bf16* __restrict__ hp) {
  int wave = (blockIdx.x * blockDim.x + threadIdx.x) >> 5;
  const int nT = HD / 16, mT = (BB * TT) / 16;
  if (wave >= nT * mT) return;
  int mTile = wave / nT, nTile = wave - mTile * nT;
  int mBase = mTile * 16, nBase = nTile * 16;
  v8f acc = wmma_tile_bf16(h_bf, ENCD, WhT, ENCD, ENCD, mBase, nBase);
  int lane = threadIdx.x & 31, hf = lane >> 4, l16 = lane & 15;
  int col = nBase + l16;
  float bias = attn_b[col];
#pragma unroll
  for (int r = 0; r < 8; ++r)
    hp[(size_t)(mBase + r + hf * 8) * HD + col] = (bf16)(acc[r] + bias);
}

// ------------------------------- per-step kernels --------------------------
// sWs = s @ attn_Ws   (s = first HD columns of xg)
__global__ void k_sws(const bf16* __restrict__ xg, const bf16* __restrict__ WsT,
                      float* __restrict__ sWs) {
  int wave = (blockIdx.x * blockDim.x + threadIdx.x) >> 5;
  const int nT = HD / 16;
  if (wave >= nT * 2) return;
  int nTile = wave >> 1, mTile = wave & 1;
  int mBase = mTile * 16, nBase = nTile * 16;
  v8f acc = wmma_tile_bf16(xg, KX, WsT, HD, HD, mBase, nBase);
  int lane = threadIdx.x & 31, hf = lane >> 4, l16 = lane & 15;
  int col = nBase + l16;
#pragma unroll
  for (int r = 0; r < 8; ++r)
    sWs[(mBase + r + hf * 8) * HD + col] = acc[r];
}

// e[b,t] = v . tanh(h_proj[b,t,:] + sWs[b,:]) ; masked by seq_lens
__global__ void k_energy(const bf16* __restrict__ hp, const float* __restrict__ sWs,
                         const float* __restrict__ v, const int* __restrict__ seq_lens,
                         float* __restrict__ e) {
  __shared__ float red[128];
  int idx = blockIdx.x;                 // b*TT + t
  int b = idx / TT, t = idx - b * TT;
  const bf16*  row = hp  + (size_t)idx * HD;
  const float* sw  = sWs + b * HD;
  float p = 0.f;
  for (int h = threadIdx.x; h < HD; h += 128)
    p += v[h] * tanhf((float)row[h] + sw[h]);
  red[threadIdx.x] = p;
  __syncthreads();
  for (int s = 64; s > 0; s >>= 1) {
    if (threadIdx.x < s) red[threadIdx.x] += red[threadIdx.x + s];
    __syncthreads();
  }
  if (threadIdx.x == 0) e[idx] = (t < seq_lens[b]) ? red[0] : NEGINF;
}

// in-place softmax over T per batch row
__global__ void k_softmax(float* __restrict__ e) {
  __shared__ float red[512];
  int b = blockIdx.x, tid = threadIdx.x;
  float val = (tid < TT) ? e[b * TT + tid] : NEGINF;
  red[tid] = val; __syncthreads();
  for (int s = 256; s > 0; s >>= 1) {
    if (tid < s) red[tid] = fmaxf(red[tid], red[tid + s]);
    __syncthreads();
  }
  float m = red[0]; __syncthreads();
  float ex = (tid < TT) ? __expf(val - m) : 0.f;
  red[tid] = ex; __syncthreads();
  for (int s = 256; s > 0; s >>= 1) {
    if (tid < s) red[tid] += red[tid + s];
    __syncthreads();
  }
  float inv = 1.f / red[0];
  if (tid < TT) e[b * TT + tid] = ex * inv;
}

// g[b,:] = alpha[b,:] @ h_batch[b]; also writes bf16 g into xg[:, HD:]
__global__ void k_ctx(const float* __restrict__ alpha, const bf16* __restrict__ h_bf,
                      float* __restrict__ g, bf16* __restrict__ xg) {
  int b = blockIdx.y;
  int col = blockIdx.x * 128 + threadIdx.x;
  const float* a  = alpha + b * TT;
  const bf16*  hb = h_bf + (size_t)b * TT * ENCD + col;
  float acc = 0.f;
#pragma unroll 4
  for (int t = 0; t < TT; ++t) acc += a[t] * (float)hb[(size_t)t * ENCD];
  g[b * ENCD + col] = acc;
  xg[b * KX + HD + col] = (bf16)acc;
}

// dec = tanh( [s|g] @ [W_sy;W_gy] + b_sy + b_gy )   -> bf16
__global__ void k_dec(const bf16* __restrict__ xg, const bf16* __restrict__ WdT,
                      const float* __restrict__ b_sy, const float* __restrict__ b_gy,
                      bf16* __restrict__ dec) {
  int wave = (blockIdx.x * blockDim.x + threadIdx.x) >> 5;
  const int nT = HD / 16;
  if (wave >= nT * 2) return;
  int nTile = wave >> 1, mTile = wave & 1;
  int mBase = mTile * 16, nBase = nTile * 16;
  v8f acc = wmma_tile_bf16(xg, KX, WdT, KX, KX, mBase, nBase);
  int lane = threadIdx.x & 31, hf = lane >> 4, l16 = lane & 15;
  int col = nBase + l16;
  float bias = b_sy[col] + b_gy[col];
#pragma unroll
  for (int r = 0; r < 8; ++r)
    dec[(mBase + r + hf * 8) * HD + col] = (bf16)tanhf(acc[r] + bias);
}

// y = dec @ W_yy + b_yy  -> d_out[b, l, :]
__global__ void k_vocab(const bf16* __restrict__ dec, const bf16* __restrict__ WyyT,
                        const float* __restrict__ b_yy, float* __restrict__ out, int l) {
  int wave = (blockIdx.x * blockDim.x + threadIdx.x) >> 5;
  const int nT = VSZ / 16;      // 625
  if (wave >= nT * 2) return;
  int nTile = wave >> 1, mTile = wave & 1;
  int mBase = mTile * 16, nBase = nTile * 16;
  v8f acc = wmma_tile_bf16(dec, HD, WyyT, HD, HD, mBase, nBase);
  int lane = threadIdx.x & 31, hf = lane >> 4, l16 = lane & 15;
  int col = nBase + l16;
  float bias = b_yy[col];
#pragma unroll
  for (int r = 0; r < 8; ++r) {
    int b = mBase + r + hf * 8;
    out[(size_t)b * (LL * VSZ) + (size_t)l * VSZ + col] = acc[r] + bias;
  }
}

// rec = [s|g] @ [W_sr;W_gr] + b_sr + b_gr + E_yr[labels[b,l]]
__global__ void k_rec(const bf16* __restrict__ xg, const bf16* __restrict__ WrT,
                      const float* __restrict__ b_sr, const float* __restrict__ b_gr,
                      const float* __restrict__ E_yr, const int* __restrict__ labels,
                      float* __restrict__ rec, int l) {
  int wave = (blockIdx.x * blockDim.x + threadIdx.x) >> 5;
  const int nT = H4 / 16;       // 128
  if (wave >= nT * 2) return;
  int nTile = wave >> 1, mTile = wave & 1;
  int mBase = mTile * 16, nBase = nTile * 16;
  v8f acc = wmma_tile_bf16(xg, KX, WrT, KX, KX, mBase, nBase);
  int lane = threadIdx.x & 31, hf = lane >> 4, l16 = lane & 15;
  int col = nBase + l16;
  float bias = b_sr[col] + b_gr[col];
#pragma unroll
  for (int r = 0; r < 8; ++r) {
    int b = mBase + r + hf * 8;
    int lbl = labels[b * LL + l];
    rec[b * H4 + col] = acc[r] + bias + E_yr[(size_t)lbl * H4 + col];
  }
}

// LSTM pointwise; writes new s (bf16) into xg[:, :HD]
__global__ void k_lstm(const float* __restrict__ rec, float* __restrict__ c,
                       bf16* __restrict__ xg) {
  int b = blockIdx.x, h = threadIdx.x;
  const float* rp = rec + b * H4;
  float iv = rp[h], fv = rp[HD + h], gv = rp[2 * HD + h], ov = rp[3 * HD + h];
  float cn = sigmoidf(fv) * c[b * HD + h] + sigmoidf(iv) * tanhf(gv);
  float sn = sigmoidf(ov) * tanhf(cn);
  c[b * HD + h] = cn;
  xg[b * KX + h] = (bf16)sn;
}

// ---------------------------------------------------------------------------
extern "C" void kernel_launch(void* const* d_in, const int* in_sizes, int n_in,
                              void* d_out, int out_size, void* d_ws, size_t ws_size,
                              hipStream_t stream) {
  (void)in_sizes; (void)n_in; (void)out_size;
  const float* h_batch = (const float*)d_in[0];
  const int*   seq_lens= (const int*)  d_in[1];
  const int*   labels  = (const int*)  d_in[2];
  const float* attn_Ws = (const float*)d_in[3];
  const float* attn_Wh = (const float*)d_in[4];
  const float* attn_b  = (const float*)d_in[5];
  const float* attn_v  = (const float*)d_in[6];
  const float* W_sy    = (const float*)d_in[7];
  const float* b_sy    = (const float*)d_in[8];
  const float* W_gy    = (const float*)d_in[9];
  const float* b_gy    = (const float*)d_in[10];
  const float* W_yy    = (const float*)d_in[11];
  const float* b_yy    = (const float*)d_in[12];
  const float* E_yr    = (const float*)d_in[13];
  const float* W_sr    = (const float*)d_in[14];
  const float* b_sr    = (const float*)d_in[15];
  const float* W_gr    = (const float*)d_in[16];
  const float* b_gr    = (const float*)d_in[17];
  float* out = (float*)d_out;

  // ---- workspace layout (256B aligned) ----
  char* base = (char*)d_ws;
  size_t off = 0;
  auto alloc = [&](size_t bytes) -> char* {
    char* p = base + off;
    off = (off + bytes + 255) & ~(size_t)255;
    return p;
  };
  bf16*  h_bf  = (bf16*) alloc((size_t)BB * TT * ENCD * 2);
  bf16*  hp    = (bf16*) alloc((size_t)BB * TT * HD * 2);
  bf16*  WhT   = (bf16*) alloc((size_t)HD * ENCD * 2);
  bf16*  WsT   = (bf16*) alloc((size_t)HD * HD * 2);
  bf16*  WdT   = (bf16*) alloc((size_t)HD * KX * 2);
  bf16*  WyyT  = (bf16*) alloc((size_t)VSZ * HD * 2);
  bf16*  WrT   = (bf16*) alloc((size_t)H4 * KX * 2);
  bf16*  xg    = (bf16*) alloc((size_t)BB * KX * 2);
  bf16*  dec   = (bf16*) alloc((size_t)BB * HD * 2);
  float* sWs   = (float*)alloc((size_t)BB * HD * 4);
  float* eAl   = (float*)alloc((size_t)BB * TT * 4);
  float* g     = (float*)alloc((size_t)BB * ENCD * 4);
  float* rec   = (float*)alloc((size_t)BB * H4 * 4);
  float* c     = (float*)alloc((size_t)BB * HD * 4);
  if (off > ws_size) return;   // workspace too small — cannot proceed

  auto cdiv = [](int a, int b) { return (a + b - 1) / b; };

  // ---- one-time setup: bf16 conversion + weight transposes ----
  { int n = BB * TT * ENCD; k_cvt<<<cdiv(n,256),256,0,stream>>>(h_batch, h_bf, n); }
  k_cvtT<<<cdiv(ENCD*HD,256),256,0,stream>>>(attn_Wh, WhT, ENCD, HD, ENCD, 0);
  k_cvtT<<<cdiv(HD*HD,256),256,0,stream>>>(attn_Ws, WsT, HD, HD, HD, 0);
  k_cvtT<<<cdiv(HD*HD,256),256,0,stream>>>(W_sy, WdT, HD, HD, KX, 0);
  k_cvtT<<<cdiv(ENCD*HD,256),256,0,stream>>>(W_gy, WdT, ENCD, HD, KX, HD);
  k_cvtT<<<cdiv(HD*VSZ,256),256,0,stream>>>(W_yy, WyyT, HD, VSZ, HD, 0);
  k_cvtT<<<cdiv(HD*H4,256),256,0,stream>>>(W_sr, WrT, HD, H4, KX, 0);
  k_cvtT<<<cdiv(ENCD*H4,256),256,0,stream>>>(W_gr, WrT, ENCD, H4, KX, HD);
  k_init<<<cdiv(BB*KX,256),256,0,stream>>>(c, xg);

  // h_proj GEMM: (BB*TT/16) * (HD/16) = 32000 tiles, 8 waves/block
  k_hproj<<<cdiv((BB*TT/16)*(HD/16),8),256,0,stream>>>(h_bf, WhT, attn_b, hp);

  // ---- sequential decode loop ----
  const int tiles_sws = (HD / 16) * 2;    // 64
  const int tiles_dec = (HD / 16) * 2;    // 64
  const int tiles_y   = (VSZ / 16) * 2;   // 1250
  const int tiles_rec = (H4 / 16) * 2;    // 256
  for (int l = 0; l < LL; ++l) {
    k_sws    <<<cdiv(tiles_sws,8),256,0,stream>>>(xg, WsT, sWs);
    k_energy <<<BB*TT,128,0,stream>>>(hp, sWs, attn_v, seq_lens, eAl);
    k_softmax<<<BB,512,0,stream>>>(eAl);
    k_ctx    <<<dim3(ENCD/128, BB),128,0,stream>>>(eAl, h_bf, g, xg);
    k_dec    <<<cdiv(tiles_dec,8),256,0,stream>>>(xg, WdT, b_sy, b_gy, dec);
    k_vocab  <<<cdiv(tiles_y,8),256,0,stream>>>(dec, WyyT, b_yy, out, l);
    k_rec    <<<cdiv(tiles_rec,8),256,0,stream>>>(xg, WrT, b_sr, b_gr, E_yr, labels, rec, l);
    k_lstm   <<<BB,HD,0,stream>>>(rec, c, xg);
  }
}
